// Adapter_37787122270958
// MI455X (gfx1250) — matrix operand order repaired and checked by reading, not compiled, not run on hardware
//
#include <hip/hip_runtime.h>
#include <hip/hip_bf16.h>
#include <stdint.h>
#include <stddef.h>

#define B_SZ    2048
#define SIZE_   4096
#define DOWN_   256
#define NTASK   16
#define MAXROWS (B_SZ + NTASK * 16)   // 2304 padded order slots
#define MAXTILES (MAXROWS / 16)       // 144
#define ACT_STRIDE (DOWN_ + 8)        // bf16 elements, pad to dodge LDS bank aliasing
#define XROWS   (B_SZ + 16)           // +16 zero rows for padded slots

typedef __attribute__((ext_vector_type(16))) __bf16 v16bf;
typedef __attribute__((ext_vector_type(8)))  float  v8f;
typedef __attribute__((ext_vector_type(4)))  int    i4x;   // POD 16-byte vector
typedef __attribute__((ext_vector_type(2)))  unsigned int u2x; // POD 8-byte vector

union Frag {
  v16bf v;
  i4x   q[2];
  unsigned short h[16];
};

__device__ __forceinline__ unsigned short f2bf(float x) {
  unsigned int u = __float_as_uint(x);
  u += 0x7fffu + ((u >> 16) & 1u);   // round-to-nearest-even
  return (unsigned short)(u >> 16);
}

// ---------------------------------------------------------------------------
// Kernel 1: bucket samples by task into 16-aligned segments (padded with -1).
// ---------------------------------------------------------------------------
__global__ void adapter_bucket_kernel(const int* __restrict__ task_id,
                                      int* __restrict__ order,
                                      int* __restrict__ tileTask,
                                      int* __restrict__ numTiles) {
  __shared__ int cnt[NTASK];
  __shared__ int segStart[NTASK];
  __shared__ int fill[NTASK];
  int tid = threadIdx.x;
  if (tid < NTASK) { cnt[tid] = 0; fill[tid] = 0; }
  __syncthreads();
  for (int b = tid; b < B_SZ; b += blockDim.x)
    atomicAdd(&cnt[task_id[b]], 1);
  __syncthreads();
  if (tid == 0) {
    int off = 0, nt = 0;
    for (int t = 0; t < NTASK; ++t) {
      segStart[t] = off;
      int tiles = (cnt[t] + 15) >> 4;
      for (int i = 0; i < tiles; ++i) tileTask[nt + i] = t;
      nt  += tiles;
      off += tiles * 16;
    }
    numTiles[0] = nt;
  }
  __syncthreads();
  for (int i = tid; i < MAXROWS; i += blockDim.x) order[i] = -1;
  __syncthreads();
  for (int b = tid; b < B_SZ; b += blockDim.x) {
    int t = task_id[b];
    int pos = segStart[t] + atomicAdd(&fill[t], 1);
    order[pos] = b;
  }
}

// ---------------------------------------------------------------------------
// Kernel 2: fp32 [R][C] -> bf16 transposed [C][R], per task (blockIdx.z).
// Makes WMMA B-fragments contiguous-in-K (per-lane 2x b128 loads).
// ---------------------------------------------------------------------------
__global__ void adapter_transpose_cvt_kernel(const float* __restrict__ in,
                                             unsigned short* __restrict__ out,
                                             int R, int C) {
  __shared__ float tile[32][33];
  size_t t = blockIdx.z;
  in  += t * (size_t)R * C;
  out += t * (size_t)R * C;
  int c0 = blockIdx.x * 32;
  int r0 = blockIdx.y * 32;
  int tx = threadIdx.x, ty = threadIdx.y;
#pragma unroll
  for (int j = 0; j < 4; ++j)
    tile[ty + 8 * j][tx] = in[(size_t)(r0 + ty + 8 * j) * C + c0 + tx];
  __syncthreads();
#pragma unroll
  for (int j = 0; j < 4; ++j)
    out[(size_t)(c0 + ty + 8 * j) * R + r0 + tx] = f2bf(tile[tx][ty + 8 * j]);
}

// ---------------------------------------------------------------------------
// Kernel 2b: flat fp32 -> bf16 copy of x, plus 16 trailing zero rows
// (dummy rows of padded tiles point at row B_SZ -> branch-free inner loop).
// ---------------------------------------------------------------------------
__global__ void adapter_xcvt_kernel(const float* __restrict__ x,
                                    unsigned short* __restrict__ xbf) {
  size_t i = ((size_t)blockIdx.x * blockDim.x + threadIdx.x) * 4;
  u2x pk;
  if (i < (size_t)B_SZ * SIZE_) {
    float4 f = *(const float4*)(x + i);
    pk.x = (unsigned int)f2bf(f.x) | ((unsigned int)f2bf(f.y) << 16);
    pk.y = (unsigned int)f2bf(f.z) | ((unsigned int)f2bf(f.w) << 16);
  } else {
    pk.x = 0u; pk.y = 0u;
  }
  *(u2x*)(xbf + i) = pk;
}

// ---------------------------------------------------------------------------
// Kernel 3: one 16-sample task-uniform tile per block; 8 waves.
// down(16x256, K=4096) -> silu -> LDS bf16 -> up(16x4096, K=256) + residual.
// ---------------------------------------------------------------------------
__global__ __launch_bounds__(256) void adapter_main_kernel(
    const float* __restrict__ x,
    const unsigned short* __restrict__ xbf,   // [XROWS][SIZE] bf16
    const float* __restrict__ bd,
    const float* __restrict__ bu,
    const unsigned short* __restrict__ WdT,   // [T][DOWN][SIZE] bf16 (k contiguous)
    const unsigned short* __restrict__ WuT,   // [T][SIZE][DOWN] bf16 (h contiguous)
    const int* __restrict__ order,
    const int* __restrict__ tileTask,
    const int* __restrict__ numTiles,
    float* __restrict__ out) {
  __shared__ __align__(16) unsigned short actlds[16 * ACT_STRIDE];
  __shared__ int rows[16];
  __shared__ int s_task;

  int tile = blockIdx.x;
  if (tile >= numTiles[0]) return;          // block-uniform exit
  int tid = threadIdx.x;
  if (tid < 16) rows[tid] = order[tile * 16 + tid];
  if (tid == 0) s_task = tileTask[tile];
  __syncthreads();

  const int task = s_task;
  const int lane = tid & 31;
  const int wave = tid >> 5;
  const int nl   = lane & 15;               // N index within 16x16 tile (B/C/D)
  const int half = lane >> 4;               // lane half selects K/M sub-ranges
  const int m    = nl;                      // A-fragment row for this lane
  const int arow = rows[m];
  // dummy rows -> zeroed row B_SZ: no divergence in the K loops
  const unsigned short* xbrow =
      xbf + (size_t)(arow >= 0 ? arow : B_SZ) * SIZE_;

  const unsigned short* wd = WdT + (size_t)task * DOWN_ * SIZE_;
  const unsigned short* wu = WuT + (size_t)task * SIZE_ * DOWN_;

  // ---------------- down projection: 16 N-tiles, 2 per wave ----------------
  {
    const int n0 = (wave * 2) * 16 + nl;
    const int n1 = n0 + 16;
    const unsigned short* wcol0 = wd + (size_t)n0 * SIZE_;
    const unsigned short* wcol1 = wd + (size_t)n1 * SIZE_;
    v8f acc0 = {};
    v8f acc1 = {};
    for (int kk = 0; kk < SIZE_; kk += 32) {
      Frag a, b0, b1;
      // A: lane half h holds K = kk+8h..+7 and K = kk+16+8h..+7 (bf16, contiguous)
      const unsigned short* ap = xbrow + kk + 8 * half;
      a.q[0] = *(const i4x*)(ap);
      a.q[1] = *(const i4x*)(ap + 16);
      // B: lane half h holds K = kk+8h..+7 and K = kk+16+8h..+7 (same as A)
      const unsigned short* bp0 = wcol0 + kk + 8 * half;
      b0.q[0] = *(const i4x*)(bp0);
      b0.q[1] = *(const i4x*)(bp0 + 16);
      const unsigned short* bp1 = wcol1 + kk + 8 * half;
      b1.q[0] = *(const i4x*)(bp1);
      b1.q[1] = *(const i4x*)(bp1 + 16);
      acc0 = __builtin_amdgcn_wmma_f32_16x16x32_bf16(
          false, a.v, false, b0.v, (short)0, acc0, false, false);
      acc1 = __builtin_amdgcn_wmma_f32_16x16x32_bf16(
          false, a.v, false, b1.v, (short)0, acc1, false, false);
    }
    float bias0 = bd[task * DOWN_ + n0];
    float bias1 = bd[task * DOWN_ + n1];
#pragma unroll
    for (int r = 0; r < 8; ++r) {
      int mm = r + 8 * half;                       // C/D: row = r + 8*half
      float v0 = acc0[r] + bias0;
      float v1 = acc1[r] + bias1;
      actlds[mm * ACT_STRIDE + n0] = f2bf(v0 / (1.0f + __expf(-v0)));  // silu
      actlds[mm * ACT_STRIDE + n1] = f2bf(v1 / (1.0f + __expf(-v1)));
    }
  }
  __syncthreads();

  // ---------------- up projection: 256 N-tiles, 32 per wave ----------------
  // Hoist the whole A operand (K=256 -> 8 fragments, 64 VGPRs) into registers.
  Frag aup[8];
#pragma unroll
  for (int k8 = 0; k8 < 8; ++k8) {
    const unsigned short* ap = actlds + m * ACT_STRIDE + k8 * 32 + 8 * half;
    aup[k8].q[0] = *(const i4x*)(ap);
    aup[k8].q[1] = *(const i4x*)(ap + 16);
  }

  for (int s = 0; s < 32; ++s) {
    int nt = wave * 32 + s;
    int d  = nt * 16 + nl;
    const unsigned short* wcol = wu + (size_t)d * DOWN_;
    v8f acc = {};
#pragma unroll
    for (int k8 = 0; k8 < 8; ++k8) {
      Frag b;
      const unsigned short* bp = wcol + k8 * 32 + 8 * half;
      b.q[0] = *(const i4x*)(bp);
      b.q[1] = *(const i4x*)(bp + 16);
      acc = __builtin_amdgcn_wmma_f32_16x16x32_bf16(
          false, aup[k8].v, false, b.v, (short)0, acc, false, false);
    }
    float bias = bu[(size_t)task * SIZE_ + d];
#pragma unroll
    for (int r = 0; r < 8; ++r) {
      int mm = r + 8 * half;
      int rr = rows[mm];
      if (rr >= 0) {
        size_t idx = (size_t)rr * SIZE_ + d;
        out[idx] = x[idx] + acc[r] + bias;         // fused residual (fp32)
      }
    }
  }
}

// ---------------------------------------------------------------------------
// Host entry. Inputs: x, task_id, Wd, bd, Wu, bu (fp32 / int32).
// Workspace layout:
//   [0]            numTiles (int)
//   [256]          tileTask (<=144 ints)
//   [2048]         order    (2304 ints)
//   [65536]        WdT bf16 : 16*256*4096 elements (32 MB)
//   [+32 MB]       WuT bf16 : 16*4096*256 elements (32 MB)
//   [+64 MB]       xbf bf16 : (2048+16)*4096 elements (~16.1 MB)
// ---------------------------------------------------------------------------
extern "C" void kernel_launch(void* const* d_in, const int* in_sizes, int n_in,
                              void* d_out, int out_size, void* d_ws, size_t ws_size,
                              hipStream_t stream) {
  (void)in_sizes; (void)n_in; (void)out_size; (void)ws_size;
  const float* x       = (const float*)d_in[0];
  const int*   task_id = (const int*)d_in[1];
  const float* Wd      = (const float*)d_in[2];
  const float* bd      = (const float*)d_in[3];
  const float* Wu      = (const float*)d_in[4];
  const float* bu      = (const float*)d_in[5];
  float* out = (float*)d_out;

  char* ws = (char*)d_ws;
  int* numTiles = (int*)ws;
  int* tileTask = (int*)(ws + 256);
  int* order    = (int*)(ws + 2048);
  unsigned short* WdT = (unsigned short*)(ws + 65536);
  unsigned short* WuT = WdT + (size_t)NTASK * DOWN_ * SIZE_;
  unsigned short* xbf = WuT + (size_t)NTASK * SIZE_ * DOWN_;

  adapter_bucket_kernel<<<1, 256, 0, stream>>>(task_id, order, tileTask, numTiles);

  // Wd: [T][4096][256] -> WdT [T][256][4096]
  adapter_transpose_cvt_kernel<<<dim3(DOWN_ / 32, SIZE_ / 32, NTASK),
                                 dim3(32, 8), 0, stream>>>(Wd, WdT, SIZE_, DOWN_);
  // Wu: [T][256][4096] -> WuT [T][4096][256]
  adapter_transpose_cvt_kernel<<<dim3(SIZE_ / 32, DOWN_ / 32, NTASK),
                                 dim3(32, 8), 0, stream>>>(Wu, WuT, DOWN_, SIZE_);

  // x fp32 -> bf16 (+16 zero rows)
  {
    size_t quads = (size_t)XROWS * SIZE_ / 4;
    int blocks = (int)((quads + 255) / 256);
    adapter_xcvt_kernel<<<blocks, 256, 0, stream>>>(x, xbf);
  }

  adapter_main_kernel<<<MAXTILES, 256, 0, stream>>>(
      x, xbf, bd, bu, WdT, WuT, order, tileTask, numTiles, out);
}
